// Many2OneRNN_13529146982528
// MI455X (gfx1250) — matrix operand order repaired and checked by reading, not compiled, not run on hardware
//
#include <hip/hip_runtime.h>

typedef _Float16 v16h __attribute__((ext_vector_type(16)));
typedef _Float16 v8h  __attribute__((ext_vector_type(8)));
typedef float    v8f  __attribute__((ext_vector_type(8)));

#define B_ 64
#define S_ 4096
#define I_ 256
#define H_ 256
#define O_ 128
#define PITCH 264   // f16 elements per LDS row (256 + 8 pad -> 528B row stride, conflict-free b128 reads)

union AFrag { v16h v; v8h h[2]; };

// ---------------------------------------------------------------------------
// Kernel 1: xp[s][b][h] = sum_i x[b][s][i] * Wax_w[h][i] + Wax_b[h]
// 512 threads = 16 waves; wave w owns H-columns [16w,16w+16).
// B-fragments (Wax^T) live in VGPRs (f16); x tiles staged through LDS.
// Memory-bound: 268MB read + 268MB written ~ 23us floor at 23.3 TB/s.
// ---------------------------------------------------------------------------
__global__ __launch_bounds__(512) void rnn_xproj_wmma(
    const float* __restrict__ x, const float* __restrict__ Wax_w,
    const float* __restrict__ Wax_b, float* __restrict__ xp)
{
    __shared__ __align__(16) _Float16 xLds[16 * PITCH];
    const int tid  = threadIdx.x;
    const int w    = tid >> 5;
    const int lane = tid & 31;
    const int ln   = lane & 15;
    const int hi   = lane >> 4;
    const int n    = w * 16 + ln;          // H column owned by this lane

    // Preload B fragments: B[k][n] = Wax_w[n][k], 16 consecutive k per lane per chunk.
    v16h Bf[8];
    #pragma unroll
    for (int c = 0; c < 8; ++c) {
        const float* src = Wax_w + (size_t)n * I_ + c * 32 + hi * 16;
        v16h f;
        #pragma unroll
        for (int i = 0; i < 16; ++i) f[i] = (_Float16)src[i];
        Bf[c] = f;
    }
    const float bias = Wax_b[n];

    const int nTiles = (B_ * S_) / 16;     // 16384 row tiles
    for (int rt = blockIdx.x; rt < nTiles; rt += gridDim.x) {
        const size_t r0 = (size_t)rt * 16;
        // Cooperative stage: 512 threads x 8 f32 -> 16x256 f16 tile in LDS.
        {
            const int flat = tid * 8;
            const int row  = flat >> 8;
            const int col  = flat & 255;
            const float* src = x + (r0 + row) * (size_t)I_ + col;
            v8h d;
            #pragma unroll
            for (int i = 0; i < 8; ++i) d[i] = (_Float16)src[i];
            *(v8h*)&xLds[row * PITCH + col] = d;
        }
        __syncthreads();

        v8f acc;
        #pragma unroll
        for (int r = 0; r < 8; ++r) acc[r] = bias;

        #pragma unroll
        for (int c = 0; c < 8; ++c) {
            AFrag a;
            const int base = ln * PITCH + c * 32 + hi * 16;
            a.h[0] = *(const v8h*)&xLds[base];
            a.h[1] = *(const v8h*)&xLds[base + 8];
            acc = __builtin_amdgcn_wmma_f32_16x16x32_f16(
                false, a.v, false, Bf[c], (short)0, acc, false, false);
        }

        // Scatter to xp[s][b][h]  (rows r0..r0+15 share one batch b since S%16==0)
        const int bb = (int)(r0 >> 12);
        const int s0 = (int)(r0 & (S_ - 1));
        #pragma unroll
        for (int r = 0; r < 8; ++r) {
            const int M = r + hi * 8;
            xp[(size_t)(s0 + M) * (B_ * H_) + (size_t)bb * H_ + n] = acc[r];
        }
        __syncthreads();   // protect xLds before next stage
    }
}

// ---------------------------------------------------------------------------
// Kernel 2: sequential scan + output head (latency-bound on recurrence).
// 4 blocks (16 batch rows each, batches independent -> no inter-WG sync),
// 512 threads = 16 waves; wave w owns H-cols [16w,16w+16).
// Waa^T B-fragments held in VGPRs for the entire scan. Hidden state is
// DOUBLE-BUFFERED in LDS so each step needs only ONE barrier: step t reads
// buf[t&1] (reads are dscnt-complete before the barrier since WMMA consumes
// them) and writes buf[(t+1)&1], so there is no WAR hazard to protect.
// xp slab for step t+1 is prefetched during step t (independent of the chain).
// ---------------------------------------------------------------------------
__global__ __launch_bounds__(512) void rnn_scan_wmma(
    const float* __restrict__ xp, const float* __restrict__ Waa_w,
    const float* __restrict__ Waa_b, const float* __restrict__ Wya_w,
    const float* __restrict__ Wya_b, float* __restrict__ out)
{
    __shared__ __align__(16) _Float16 aLds[2][16 * PITCH];
    const int tid  = threadIdx.x;
    const int w    = tid >> 5;
    const int lane = tid & 31;
    const int ln   = lane & 15;
    const int hi   = lane >> 4;
    const int b0   = blockIdx.x * 16;      // batch rows owned by this block
    const int n    = w * 16 + ln;          // H column owned by this lane

    // Preload B fragments of Waa^T: B[k][n] = Waa_w[n][k].
    v16h Bf[8];
    #pragma unroll
    for (int c = 0; c < 8; ++c) {
        const float* src = Waa_w + (size_t)n * H_ + c * 32 + hi * 16;
        v16h f;
        #pragma unroll
        for (int i = 0; i < 16; ++i) f[i] = (_Float16)src[i];
        Bf[c] = f;
    }
    const float bias = Waa_b[n];

    // a_0 = 0 into buffer 0
    for (int i = tid; i < 16 * PITCH; i += 512) aLds[0][i] = (_Float16)0.f;
    __syncthreads();

    // Preload xp tile for t = 0 (C layout: VGPR r -> batch row r + hi*8).
    float xpv[8];
    {
        const float* p = xp + (size_t)b0 * H_ + n;
        #pragma unroll
        for (int r = 0; r < 8; ++r) xpv[r] = p[(r + hi * 8) * H_];
    }

    for (int t = 0; t < S_; ++t) {
        const _Float16* __restrict__ cur = aLds[t & 1];
        _Float16* __restrict__ nxt       = aLds[(t + 1) & 1];

        v8f acc;
        #pragma unroll
        for (int r = 0; r < 8; ++r) acc[r] = xpv[r] + bias;

        // Prefetch next step's xp slab (independent of recurrence).
        if (t + 1 < S_) {
            const float* p = xp + (size_t)(t + 1) * (B_ * H_) + (size_t)b0 * H_ + n;
            #pragma unroll
            for (int r = 0; r < 8; ++r) xpv[r] = p[(r + hi * 8) * H_];
        }

        // acc += a_t @ Waa^T   (8 chained f16 WMMAs, A from LDS buffer t&1)
        #pragma unroll
        for (int c = 0; c < 8; ++c) {
            AFrag a;
            const int base = ln * PITCH + c * 32 + hi * 16;
            a.h[0] = *(const v8h*)&cur[base];
            a.h[1] = *(const v8h*)&cur[base + 8];
            acc = __builtin_amdgcn_wmma_f32_16x16x32_f16(
                false, a.v, false, Bf[c], (short)0, acc, false, false);
        }

        // a_{t+1} = tanh(acc) into the OTHER buffer (safe: e->inf => 1, e->0 => -1)
        #pragma unroll
        for (int r = 0; r < 8; ++r) {
            const float e  = __expf(2.f * acc[r]);
            const float th = 1.f - 2.f / (e + 1.f);
            nxt[(r + hi * 8) * PITCH + n] = (_Float16)th;
        }
        __syncthreads();   // single barrier per step: publish a_{t+1}
    }

    // Output head: out = sigmoid(a_S @ Wya^T + b). a_S is in buffer (S_ & 1)==0.
    // O=128 -> waves 0..7 only (wave-uniform branch: EXEC all ones, WMMA-legal).
    if (w < 8) {
        const _Float16* __restrict__ fin = aLds[S_ & 1];
        const int no = w * 16 + ln;        // output column
        v8f acc;
        const float by = Wya_b[no];
        #pragma unroll
        for (int r = 0; r < 8; ++r) acc[r] = by;

        #pragma unroll
        for (int c = 0; c < 8; ++c) {
            const float* src = Wya_w + (size_t)no * H_ + c * 32 + hi * 16;
            v16h f;
            #pragma unroll
            for (int i = 0; i < 16; ++i) f[i] = (_Float16)src[i];
            AFrag a;
            const int base = ln * PITCH + c * 32 + hi * 16;
            a.h[0] = *(const v8h*)&fin[base];
            a.h[1] = *(const v8h*)&fin[base + 8];
            acc = __builtin_amdgcn_wmma_f32_16x16x32_f16(
                false, a.v, false, f, (short)0, acc, false, false);
        }
        #pragma unroll
        for (int r = 0; r < 8; ++r) {
            const float sg = 1.f / (1.f + __expf(-acc[r]));
            out[(size_t)(b0 + r + hi * 8) * O_ + no] = sg;
        }
    }
}

// ---------------------------------------------------------------------------
extern "C" void kernel_launch(void* const* d_in, const int* in_sizes, int n_in,
                              void* d_out, int out_size, void* d_ws, size_t ws_size,
                              hipStream_t stream) {
    const float* x     = (const float*)d_in[0];
    const float* Wax_w = (const float*)d_in[1];
    const float* Wax_b = (const float*)d_in[2];
    const float* Waa_w = (const float*)d_in[3];
    const float* Waa_b = (const float*)d_in[4];
    const float* Wya_w = (const float*)d_in[5];
    const float* Wya_b = (const float*)d_in[6];
    float* out = (float*)d_out;
    float* xp  = (float*)d_ws;             // [S][B][H] f32 = 256 MiB scratch

    rnn_xproj_wmma<<<2048, 512, 0, stream>>>(x, Wax_w, Wax_b, xp);
    rnn_scan_wmma<<<B_ / 16, 512, 0, stream>>>(xp, Waa_w, Waa_b, Wya_w, Wya_b, out);
}
